// Infinity_Attention_41034117546101
// MI455X (gfx1250) — matrix operand order repaired and checked by reading, not compiled, not run on hardware
//
#include <hip/hip_runtime.h>

#define BB 4
#define HH 16
#define NN 2048
#define DD 64
#define DIMX 1024
#define INNERX 1024

typedef __attribute__((ext_vector_type(16))) __bf16 v16bf;
typedef __attribute__((ext_vector_type(8)))  float  v8f;

struct alignas(16) U4 { unsigned x, y, z, w; };
struct U8v { U4 lo, hi; };
struct alignas(16) F4 { float x, y, z, w; };

static __device__ __forceinline__ v16bf v16_from(U4 lo, U4 hi) {
  U8v t; t.lo = lo; t.hi = hi;
  return __builtin_bit_cast(v16bf, t);
}

// Native bf16 converts (v_cvt_*bf16* on gfx1250) instead of manual rounding.
static __device__ __forceinline__ unsigned short f2bf(float f) {
  return __builtin_bit_cast(unsigned short, (__bf16)f);
}
static __device__ __forceinline__ float bf2f(unsigned short s) {
  return (float)__builtin_bit_cast(__bf16, s);
}
static __device__ __forceinline__ unsigned pack2(float a, float b) {
  return (unsigned)f2bf(a) | ((unsigned)f2bf(b) << 16);
}

// WMMA A-operand (16xK slice): row-major source; lane holds row m, 8-elem
// chunks at k0+8*half and k0+16+8*half (ISA 7.12.2, 16-bit A 16x32).
static __device__ __forceinline__ v16bf load_a16(const unsigned short* base,
                                                 int stride, int m, int half, int k0) {
  const U4* p = reinterpret_cast<const U4*>(base + (size_t)m * stride + k0 + 8 * half);
  return v16_from(p[0], p[2]);
}
// WMMA B-operand: source with matrix columns as contiguous rows (i.e. the
// "B matrix transposed" in memory); lane holds column n, 8-elem chunks at
// k0+8*half and k0+16+8*half (same k order as A).
static __device__ __forceinline__ v16bf load_b16(const unsigned short* base,
                                                 int stride, int n, int half, int k0) {
  const U4* p = reinterpret_cast<const U4*>(base + (size_t)n * stride + k0 + 8 * half);
  return v16_from(p[0], p[2]);
}

static __device__ __forceinline__ v8f wmma_bf16(v16bf a, v16bf b, v8f c) {
  return __builtin_amdgcn_wmma_f32_16x16x32_bf16(false, a, false, b, (short)0, c,
                                                 false, false);
}

static __device__ __forceinline__ float bperm_f(float v, int srcLane) {
  return __int_as_float(__builtin_amdgcn_ds_bpermute(srcLane << 2, __float_as_int(v)));
}

// ---------------------------------------------------------------------------
// Kernel 1: qkv = x @ W_qkv, bf16 WMMA, fused sigma(elu+1) + per-head scatter.
// Also emits transposed copies vtb = v^T and sktb = sk^T per head so later
// kernels can feed WMMA B-operands straight from global/L2 (192MB L2 holds
// the whole working set) without LDS transpose staging.
// ---------------------------------------------------------------------------
__global__ __launch_bounds__(256) void qkv_gemm_kernel(
    const float* __restrict__ X, const float* __restrict__ W,
    unsigned short* __restrict__ qb, unsigned short* __restrict__ kb,
    unsigned short* __restrict__ vb, unsigned short* __restrict__ sqb,
    unsigned short* __restrict__ skb, unsigned short* __restrict__ vtb,
    unsigned short* __restrict__ sktb) {
  __shared__ __align__(16) unsigned short ldsA[64 * 32];
  __shared__ __align__(16) unsigned short ldsB[128 * 32];
  const int t = threadIdx.x, lane = t & 31, wv = t >> 5;
  const int half = lane >> 4, l16 = lane & 15;
  const int rbase = blockIdx.y * 64;
  const int cbase = blockIdx.x * 128;
  const int waveM = wv & 3, waveN = wv >> 2;

  v8f acc[4] = {};

  for (int k0 = 0; k0 < DIMX; k0 += 32) {
    __syncthreads();
    {  // stage A: 64 rows x 32 K, fp32 -> bf16, row-major
      int row = t >> 2, kk = (t & 3) * 8;
      const F4* src = reinterpret_cast<const F4*>(
          X + (size_t)(rbase + row) * DIMX + k0 + kk);
      if (k0 + 32 < DIMX)
        __builtin_prefetch(X + (size_t)(rbase + row) * DIMX + k0 + 32 + kk, 0, 1);
      F4 f0 = src[0], f1 = src[1];
      U4 u;
      u.x = pack2(f0.x, f0.y); u.y = pack2(f0.z, f0.w);
      u.z = pack2(f1.x, f1.y); u.w = pack2(f1.z, f1.w);
      *reinterpret_cast<U4*>(&ldsA[row * 32 + kk]) = u;
    }
    {  // stage B transposed: W[k][c] -> ldsB[c][k]
      int kk = t >> 3, c0 = (t & 7) * 16;
      const F4* src = reinterpret_cast<const F4*>(
          W + (size_t)(k0 + kk) * 3072 + cbase + c0);
      #pragma unroll
      for (int q4 = 0; q4 < 4; ++q4) {
        F4 f = src[q4];
        ldsB[(c0 + q4 * 4 + 0) * 32 + kk] = f2bf(f.x);
        ldsB[(c0 + q4 * 4 + 1) * 32 + kk] = f2bf(f.y);
        ldsB[(c0 + q4 * 4 + 2) * 32 + kk] = f2bf(f.z);
        ldsB[(c0 + q4 * 4 + 3) * 32 + kk] = f2bf(f.w);
      }
    }
    __syncthreads();
    v16bf a = load_a16(ldsA, 32, waveM * 16 + l16, half, 0);
    #pragma unroll
    for (int j = 0; j < 4; ++j) {
      v16bf b = load_b16(ldsB, 32, waveN * 64 + j * 16 + l16, half, 0);
      acc[j] = wmma_bf16(a, b, acc[j]);
    }
  }

  // scatter: col -> (which, head, d); row -> (b, n). C layout: lane = col L%16,
  // reg r -> row r + 8*(L/16).
  #pragma unroll
  for (int j = 0; j < 4; ++j) {
    int col = cbase + waveN * 64 + j * 16 + l16;
    int which = col >> 10;
    int rem = col & 1023;
    int h = rem >> 6, d = rem & 63;
    #pragma unroll
    for (int r = 0; r < 8; ++r) {
      int row = rbase + waveM * 16 + r + 8 * half;
      int b = row >> 11, n = row & 2047;
      int bh = b * HH + h;
      size_t idx  = ((size_t)bh * NN + n) * DD + d;       // [b,h,n,d]
      size_t idxT = ((size_t)bh * DD + d) * NN + n;       // [b,h,d,n]
      float v = acc[j][r];
      unsigned short raw = f2bf(v);
      float s = v > 0.f ? v + 1.f : __expf(v);  // elu(v)+1
      if (which == 0) {
        qb[idx] = raw; sqb[idx] = f2bf(s);
      } else if (which == 1) {
        unsigned short sr = f2bf(s);
        kb[idx] = raw; skb[idx] = sr; sktb[idxT] = sr;
      } else {
        vb[idx] = raw; vtb[idxT] = raw;
      }
    }
  }
}

// ---------------------------------------------------------------------------
// Kernel 2: mem branch. mode 0: outA = g * (sq@M)/(sq@Z);
//           mode 1: vmt = transpose_bf16(v - (sk@M)/(sk@Z))  [b,h,d,n]
// One WG per (b, h, 128-row block); 8 waves x 16 rows.
// ---------------------------------------------------------------------------
__global__ __launch_bounds__(256) void mem_branch_kernel(
    const unsigned short* __restrict__ src, const unsigned short* __restrict__ vb,
    const float* __restrict__ Mmat, const float* __restrict__ Zvec,
    const float* __restrict__ beta_gate, float* __restrict__ outA,
    unsigned short* __restrict__ vmt, int mode) {
  __shared__ __align__(16) unsigned short ldsMt[64 * 64];
  __shared__ float ldsZ[64];
  const int t = threadIdx.x, lane = t & 31, wv = t >> 5;
  const int half = lane >> 4, l16 = lane & 15;
  const int nblk = blockIdx.x, h = blockIdx.y, b = blockIdx.z;
  const int bh = b * HH + h;
  {  // stage M transposed: M[kd][v] -> ldsMt[v][kd]
    int kd = t >> 2, v0 = (t & 3) * 16;
    const F4* sm = reinterpret_cast<const F4*>(
        Mmat + (size_t)bh * 64 * 64 + (size_t)kd * 64 + v0);
    #pragma unroll
    for (int q4 = 0; q4 < 4; ++q4) {
      F4 f = sm[q4];
      ldsMt[(v0 + q4 * 4 + 0) * 64 + kd] = f2bf(f.x);
      ldsMt[(v0 + q4 * 4 + 1) * 64 + kd] = f2bf(f.y);
      ldsMt[(v0 + q4 * 4 + 2) * 64 + kd] = f2bf(f.z);
      ldsMt[(v0 + q4 * 4 + 3) * 64 + kd] = f2bf(f.w);
    }
    if (t < 64) ldsZ[t] = Zvec[(size_t)bh * 64 + t];
  }
  __syncthreads();

  const int wbase = nblk * 128 + wv * 16;
  const unsigned short* abase = src + ((size_t)bh * NN + wbase) * DD;
  v8f acc[4] = {};
  #pragma unroll
  for (int ks = 0; ks < 64; ks += 32) {
    v16bf a = load_a16(abase, 64, l16, half, ks);
    #pragma unroll
    for (int j = 0; j < 4; ++j) {
      v16bf bm = load_b16(ldsMt, 64, j * 16 + l16, half, ks);
      acc[j] = wmma_bf16(a, bm, acc[j]);
    }
  }
  // denominator z[row] = sigma(row) . Z ; lane L partial over d-half
  float zp = 0.f;
  {
    const unsigned short* rp = abase + (size_t)l16 * DD + half * 32;
    #pragma unroll
    for (int i = 0; i < 32; ++i) zp += bf2f(rp[i]) * ldsZ[half * 32 + i];
  }
  zp += bperm_f(zp, lane ^ 16);  // lane L (both halves) holds z for row L&15

  float g = 1.f / (1.f + __expf(-beta_gate[h]));
  #pragma unroll
  for (int j = 0; j < 4; ++j) {
    int col = j * 16 + l16;
    #pragma unroll
    for (int r = 0; r < 8; ++r) {
      int rl = r + 8 * half;
      float z = bperm_f(zp, rl);
      float val = acc[j][r] / z;
      int n = wbase + rl;
      if (mode == 0) {
        outA[((size_t)b * NN + n) * INNERX + h * 64 + col] = g * val;
      } else {
        float diff = bf2f(vb[((size_t)bh * NN + n) * DD + col]) - val;
        vmt[((size_t)bh * DD + col) * NN + n] = f2bf(diff);
      }
    }
  }
}

// ---------------------------------------------------------------------------
// Kernel 3: flash attention; outA += (1-g) * softmax(q k^T / 8) v.
// One WG per (b,h,128 q-rows); wave = 16 q rows; key chunks of 64.
// All GEMM operands stream from global/L2 (Q,K row-major; V via vtb = V^T);
// only P round-trips through per-wave LDS. No workgroup barriers.
// ---------------------------------------------------------------------------
__global__ __launch_bounds__(256) void attn_kernel(
    const unsigned short* __restrict__ qb, const unsigned short* __restrict__ kb,
    const unsigned short* __restrict__ vtb, const float* __restrict__ beta_gate,
    float* __restrict__ outA) {
  __shared__ __align__(16) unsigned short ldsP[8 * 16 * 64];
  const int t = threadIdx.x, lane = t & 31, wv = t >> 5;
  const int half = lane >> 4, l16 = lane & 15;
  const int qblk = blockIdx.x, h = blockIdx.y, b = blockIdx.z;
  const int bh = b * HH + h;
  const int qbase = qblk * 128 + wv * 16;
  const unsigned short* qrow = qb + ((size_t)bh * NN + qbase) * DD;
  const unsigned short* vtbase = vtb + (size_t)bh * DD * NN;
  unsigned short* myP = &ldsP[wv * 16 * 64];

  v8f O[4] = {};
  float mrow[8], lrow[8];
  #pragma unroll
  for (int r = 0; r < 8; ++r) { mrow[r] = -1e30f; lrow[r] = 0.f; }

  for (int jc = 0; jc < NN; jc += 64) {
    // S = q k^T : K row-major is exactly the B-operand access pattern.
    v8f S[4] = {};
    const unsigned short* krow = kb + ((size_t)bh * NN + jc) * DD;
    if (jc + 64 < NN)
      __builtin_prefetch(kb + ((size_t)bh * NN + jc + 64 + (lane << 1)) * DD, 0, 1);
    #pragma unroll
    for (int ks = 0; ks < 64; ks += 32) {
      v16bf a = load_a16(qrow, 64, l16, half, ks);
      #pragma unroll
      for (int jt = 0; jt < 4; ++jt) {
        v16bf bm = load_b16(krow, 64, jt * 16 + l16, half, ks);
        S[jt] = wmma_bf16(a, bm, S[jt]);
      }
    }

    // online softmax; rows live per (reg, lane-half), reduce across 16 lanes
    #pragma unroll
    for (int r = 0; r < 8; ++r) {
      float mx = -1e30f;
      #pragma unroll
      for (int jt = 0; jt < 4; ++jt) mx = fmaxf(mx, S[jt][r] * 0.125f);
      #pragma unroll
      for (int msk = 1; msk < 16; msk <<= 1) mx = fmaxf(mx, bperm_f(mx, lane ^ msk));
      float mnew = fmaxf(mrow[r], mx);
      float alpha = __expf(mrow[r] - mnew);
      mrow[r] = mnew;
      float rs = 0.f;
      #pragma unroll
      for (int jt = 0; jt < 4; ++jt) {
        float p = __expf(S[jt][r] * 0.125f - mnew);
        S[jt][r] = p;
        rs += p;
      }
      #pragma unroll
      for (int msk = 1; msk < 16; msk <<= 1) rs += bperm_f(rs, lane ^ msk);
      lrow[r] = lrow[r] * alpha + rs;
      #pragma unroll
      for (int dt = 0; dt < 4; ++dt) O[dt][r] *= alpha;
      int rl = r + 8 * half;
      #pragma unroll
      for (int jt = 0; jt < 4; ++jt)
        myP[rl * 64 + jt * 16 + l16] = f2bf(S[jt][r]);
    }
    asm volatile("s_wait_dscnt 0" ::: "memory");  // wave-local P RAW in LDS

    // O += P @ V  (P row-major per-wave in LDS = A operand; V^T from global)
    #pragma unroll
    for (int ks = 0; ks < 64; ks += 32) {
      v16bf a = load_a16(myP, 64, l16, half, ks);
      #pragma unroll
      for (int dt = 0; dt < 4; ++dt) {
        v16bf bm = load_b16(vtbase, NN, dt * 16 + l16, half, jc + ks);
        O[dt] = wmma_bf16(a, bm, O[dt]);
      }
    }
  }

  float g = 1.f / (1.f + __expf(-beta_gate[h]));
  float og = 1.f - g;
  #pragma unroll
  for (int dt = 0; dt < 4; ++dt) {
    int d = dt * 16 + l16;
    #pragma unroll
    for (int r = 0; r < 8; ++r) {
      int n = qbase + r + 8 * half;
      size_t idx = ((size_t)b * NN + n) * INNERX + h * 64 + d;
      outA[idx] += og * O[dt][r] / lrow[r];
    }
  }
}

// ---------------------------------------------------------------------------
// Kernel 4: M_new = beta * M + sk^T @ (v - moment). One WG (4 waves) per (b,h).
// Both operands are pre-transposed bf16 in ws (sktb, vmt) -> pure global-fed
// WMMA loop over K = 2048, no LDS at all.
// ---------------------------------------------------------------------------
__global__ __launch_bounds__(128) void state_update_kernel(
    const unsigned short* __restrict__ sktb, const unsigned short* __restrict__ vmt,
    const float* __restrict__ Mmat, const float* __restrict__ beta_lin,
    float* __restrict__ outM) {
  const int t = threadIdx.x, lane = t & 31, wv = t >> 5;  // 4 waves
  const int half = lane >> 4, l16 = lane & 15;
  const int h = blockIdx.x, b = blockIdx.y;
  const int bh = b * HH + h;
  const unsigned short* abase = sktb + ((size_t)bh * DD + wv * 16) * NN;
  const unsigned short* bbase = vmt + (size_t)bh * DD * NN;
  v8f acc[4] = {};
  for (int n0 = 0; n0 < NN; n0 += 32) {
    v16bf a = load_a16(abase, NN, l16, half, n0);
    #pragma unroll
    for (int jt = 0; jt < 4; ++jt) {
      v16bf bm = load_b16(bbase, NN, jt * 16 + l16, half, n0);
      acc[jt] = wmma_bf16(a, bm, acc[jt]);
    }
  }
  float sg = 1.f / (1.f + __expf(-beta_lin[0]));
  float beta = fminf(0.999f, fmaxf(0.9f, sg));
  #pragma unroll
  for (int jt = 0; jt < 4; ++jt) {
    int col = jt * 16 + l16;
    #pragma unroll
    for (int r = 0; r < 8; ++r) {
      int row = wv * 16 + r + 8 * half;
      size_t idx = (size_t)bh * 64 * 64 + (size_t)row * 64 + col;
      outM[idx] = beta * Mmat[idx] + acc[jt][r];
    }
  }
}

// ---------------------------------------------------------------------------
// Kernel 5: Z_new = beta * Z + sum_n sk. One WG per (b,h); sums columns of
// sk^T (sktb) so each thread reads a contiguous row.
// ---------------------------------------------------------------------------
__global__ __launch_bounds__(256) void z_update_kernel(
    const unsigned short* __restrict__ sktb, const float* __restrict__ Zvec,
    const float* __restrict__ beta_lin, float* __restrict__ outZ) {
  __shared__ float red[4][64];
  const int t = threadIdx.x;
  const int bh = blockIdx.x;
  int d = t & 63, grp = t >> 6;
  const unsigned short* row = sktb + ((size_t)bh * DD + d) * NN;
  float p = 0.f;
  for (int n = grp * 512; n < grp * 512 + 512; ++n) p += bf2f(row[n]);
  red[grp][d] = p;
  __syncthreads();
  if (t < 64) {
    float sg = 1.f / (1.f + __expf(-beta_lin[0]));
    float beta = fminf(0.999f, fmaxf(0.9f, sg));
    float s = red[0][t] + red[1][t] + red[2][t] + red[3][t];
    outZ[(size_t)bh * 64 + t] = beta * Zvec[(size_t)bh * 64 + t] + s;
  }
}

extern "C" void kernel_launch(void* const* d_in, const int* in_sizes, int n_in,
                              void* d_out, int out_size, void* d_ws, size_t ws_size,
                              hipStream_t stream) {
  (void)in_sizes; (void)n_in; (void)out_size; (void)ws_size;
  const float* x         = (const float*)d_in[0];
  const float* M         = (const float*)d_in[1];
  const float* Z         = (const float*)d_in[2];
  const float* W         = (const float*)d_in[3];
  const float* beta_lin  = (const float*)d_in[4];
  const float* beta_gate = (const float*)d_in[5];

  float* outA = (float*)d_out;
  float* outM = outA + (size_t)BB * NN * INNERX;
  float* outZ = outM + (size_t)BB * HH * DD * DD;

  const size_t E = (size_t)BB * HH * NN * DD;  // 8,388,608 elems per array
  unsigned short* qb   = (unsigned short*)d_ws;
  unsigned short* kb   = qb + E;
  unsigned short* vb   = kb + E;
  unsigned short* sqb  = vb + E;
  unsigned short* skb  = sqb + E;
  unsigned short* vtb  = skb + E;   // v^T   [b,h,d,n]
  unsigned short* sktb = vtb + E;   // sk^T  [b,h,d,n]
  unsigned short* vmt  = sktb + E;  // (v-moment)^T [b,h,d,n]
  // total ws use: 16*E bytes = 128 MiB

  qkv_gemm_kernel<<<dim3(24, 128), 256, 0, stream>>>(x, W, qb, kb, vb, sqb, skb,
                                                     vtb, sktb);
  mem_branch_kernel<<<dim3(16, HH, BB), 256, 0, stream>>>(sqb, vb, M, Z, beta_gate,
                                                          outA, vmt, 0);
  mem_branch_kernel<<<dim3(16, HH, BB), 256, 0, stream>>>(skb, vb, M, Z, beta_gate,
                                                          outA, vmt, 1);
  attn_kernel<<<dim3(16, HH, BB), 256, 0, stream>>>(qb, kb, vtb, beta_gate, outA);
  state_update_kernel<<<dim3(HH, BB), 128, 0, stream>>>(sktb, vmt, M, beta_lin, outM);
  z_update_kernel<<<BB * HH, 256, 0, stream>>>(sktb, Z, beta_lin, outZ);
}
